// Bidirectional_LSTM_Decoder_47399259079352
// MI455X (gfx1250) — matrix operand order repaired and checked by reading, not compiled, not run on hardware
//
#include <hip/hip_runtime.h>

// ---------------------------------------------------------------- types
typedef __attribute__((ext_vector_type(16))) __bf16 v16bf;
typedef __attribute__((ext_vector_type(8)))  float  v8f;

// ---------------------------------------------------------------- problem constants (from reference)
constexpr int L_ = 2;
constexpr int H_ = 512;
constexpr int D_ = 512;
constexpr int B_ = 64;
constexpr int T_ = 512;
constexpr int G_ = 4 * H_;   // 2048 gate rows

// ---------------------------------------------------------------- device helpers
__device__ __forceinline__ unsigned short f32_to_bf16_bits(float f) {
  union { float f; unsigned u; } v; v.f = f;
  unsigned u = v.u;
  unsigned r = u + 0x7fffu + ((u >> 16) & 1u);   // round-to-nearest-even
  return (unsigned short)(r >> 16);
}
__device__ __forceinline__ __bf16 f2bf(float f) {
  union { unsigned short s; __bf16 b; } o; o.s = f32_to_bf16_bits(f); return o.b;
}

// A-fragment (16x32 bf16, M=lane&15): elements 0..7 <- K off+0..7, 8..15 <- K off+16..23
// caller bakes the per-lane "+8*hi" into the row pointer.
__device__ __forceinline__ v16bf load_a_frag(const __bf16* p) {
  union { uint4 u[2]; v16bf v; } t;
  t.u[0] = *reinterpret_cast<const uint4*>(p);
  t.u[1] = *reinterpret_cast<const uint4*>(p + 16);
  return t.v;
}
// B-fragment (32x16 bf16, N=lane&15): elements 0..7 <- K off+0..7, 8..15 <- K off+16..23
__device__ __forceinline__ v16bf load_b_frag(const __bf16* p) {
  union { uint4 u[2]; v16bf v; } t;
  t.u[0] = *reinterpret_cast<const uint4*>(p);
  t.u[1] = *reinterpret_cast<const uint4*>(p + 16);
  return t.v;
}
__device__ __forceinline__ v8f wmma_bf16(v16bf a, v16bf b, v8f c) {
  return __builtin_amdgcn_wmma_f32_16x16x32_bf16(false, a, false, b, (short)0, c, false, false);
}
__device__ __forceinline__ float sigmoidf_(float x) {
  return 1.0f / (1.0f + __expf(-x));
}
__device__ __forceinline__ float tanhf_(float x) {
  float e = __expf(2.0f * x);          // inf-safe: e=inf -> 1, e=0 -> -1
  return 1.0f - 2.0f / (e + 1.0f);
}

// ---------------------------------------------------------------- prep kernels
__global__ void cvt_f32_bf16_kernel(const float* __restrict__ in, __bf16* __restrict__ out, int n) {
  int i = blockIdx.x * blockDim.x + threadIdx.x;
  if (i < n) out[i] = f2bf(in[i]);
}

__global__ void bias_sum_kernel(const float* __restrict__ a, const float* __restrict__ b,
                                float* __restrict__ out, int n) {
  int i = blockIdx.x * blockDim.x + threadIdx.x;
  if (i < n) out[i] = a[i] + b[i];
}

// h0/c0: (L, B, 2H) -> per-(layer,dir) contiguous (B,H) state; hbf buffer 0
__global__ void init_state_kernel(const float* __restrict__ h0, const float* __restrict__ c0,
                                  float* __restrict__ hs, float* __restrict__ cs,
                                  __bf16* __restrict__ hbf) {
  int idx = blockIdx.x * blockDim.x + threadIdx.x;
  if (idx >= L_ * B_ * 2 * H_) return;
  int l   = idx / (B_ * 2 * H_);
  int r   = idx - l * (B_ * 2 * H_);
  int b   = r / (2 * H_);
  int col = r - b * (2 * H_);
  int dir = col >> 9;            // H_ = 512
  int j   = col & (H_ - 1);
  size_t s   = ((size_t)(l * 2 + dir) * B_ + b) * H_ + j;
  size_t sbf = ((size_t)((l * 2 + dir) * 2 + 0)) * (B_ * H_) + (size_t)b * H_ + j;
  float hv = h0[idx];
  hs[s] = hv;
  hbf[sbf] = f2bf(hv);
  cs[s] = c0[idx];
}

// assemble final h,c outputs: (L,B,2H) from per-(layer,dir) state
__global__ void finalize_kernel(const float* __restrict__ hs, const float* __restrict__ cs,
                                float* __restrict__ outh, float* __restrict__ outc) {
  int idx = blockIdx.x * blockDim.x + threadIdx.x;
  if (idx >= L_ * B_ * 2 * H_) return;
  int l   = idx / (B_ * 2 * H_);
  int r   = idx - l * (B_ * 2 * H_);
  int b   = r / (2 * H_);
  int col = r - b * (2 * H_);
  int dir = col >> 9;
  int j   = col & (H_ - 1);
  size_t s = ((size_t)(l * 2 + dir) * B_ + b) * H_ + j;
  outh[idx] = hs[s];
  outc[idx] = cs[s];
}

// ---------------------------------------------------------------- fused LSTM step (one layer, both dirs)
// grid: 32 blocks x 256 threads = 256 waves = 2 dirs * 4 batch-tiles * 32 hidden-col-tiles.
// Each wave: 16(batch) x 16(hidden) cells; computes all 4 gate tiles with WMMA over
// K1 (x-part) + H (h-part), then does the cell update in-register.
__global__ __launch_bounds__(256) void lstm_step_kernel(
    const __bf16* __restrict__ A1f, const __bf16* __restrict__ A1b, long sA1, int K1,
    const __bf16* __restrict__ W1f, const __bf16* __restrict__ W1b,
    const __bf16* __restrict__ A2f, const __bf16* __restrict__ A2b,
    const __bf16* __restrict__ W2f, const __bf16* __restrict__ W2b,
    const float*  __restrict__ biasf, const float* __restrict__ biasb,
    float* __restrict__ csf, float* __restrict__ csb,
    float* __restrict__ hsf, float* __restrict__ hsb,
    __bf16* __restrict__ hnf, __bf16* __restrict__ hnb,
    float* __restrict__ outp, int toutf, int toutb) {
  const int lane  = threadIdx.x & 31;
  const int wid   = (blockIdx.x << 3) + (threadIdx.x >> 5);
  const int dir   = wid >> 7;            // 0 = forward, 1 = backward
  const int rem   = wid & 127;
  const int m0    = (rem & 3) << 4;      // batch tile origin
  const int n0    = (rem >> 2) << 4;     // hidden-col tile origin
  const int laneN = lane & 15;
  const int hi    = lane >> 4;

  const __bf16* A1   = dir ? A1b : A1f;
  const __bf16* W1   = dir ? W1b : W1f;
  const __bf16* A2   = dir ? A2b : A2f;
  const __bf16* W2   = dir ? W2b : W2f;
  const float*  bias = dir ? biasb : biasf;
  float*  cs   = dir ? csb : csf;
  float*  hs   = dir ? hsb : hsf;
  __bf16* hn   = dir ? hnb : hnf;
  const int tout = dir ? toutb : toutf;

  v8f acc0 = {}, acc1 = {}, acc2 = {}, acc3 = {};

  // ---- phase 1: x-part (A: K1 wide, row stride sA1; W: (4H x K1) row-major)
  {
    const __bf16* arow = A1 + (size_t)(m0 + laneN) * (size_t)sA1 + (hi << 3);
    const size_t gstr  = (size_t)H_ * (size_t)K1;
    const __bf16* wrow = W1 + (size_t)(n0 + laneN) * (size_t)K1 + (hi << 3);
#pragma unroll 4
    for (int kk = 0; kk < K1; kk += 32) {
      v16bf a = load_a_frag(arow + kk);
      acc0 = wmma_bf16(a, load_b_frag(wrow + kk),            acc0);
      acc1 = wmma_bf16(a, load_b_frag(wrow + kk + gstr),     acc1);
      acc2 = wmma_bf16(a, load_b_frag(wrow + kk + 2 * gstr), acc2);
      acc3 = wmma_bf16(a, load_b_frag(wrow + kk + 3 * gstr), acc3);
    }
  }
  // ---- phase 2: recurrent h-part (A: (B x H) contiguous; W: (4H x H) row-major)
  {
    const __bf16* arow = A2 + (size_t)(m0 + laneN) * H_ + (hi << 3);
    const size_t gstr  = (size_t)H_ * (size_t)H_;
    const __bf16* wrow = W2 + (size_t)(n0 + laneN) * H_ + (hi << 3);
#pragma unroll 4
    for (int kk = 0; kk < H_; kk += 32) {
      v16bf a = load_a_frag(arow + kk);
      acc0 = wmma_bf16(a, load_b_frag(wrow + kk),            acc0);
      acc1 = wmma_bf16(a, load_b_frag(wrow + kk + gstr),     acc1);
      acc2 = wmma_bf16(a, load_b_frag(wrow + kk + 2 * gstr), acc2);
      acc3 = wmma_bf16(a, load_b_frag(wrow + kk + 3 * gstr), acc3);
    }
  }

  // ---- cell update (C/D layout: element v -> row m0+v+8*hi, col n0+laneN)
  const int hcol = n0 + laneN;
  const float b_i = bias[0 * H_ + hcol];
  const float b_f = bias[1 * H_ + hcol];
  const float b_g = bias[2 * H_ + hcol];
  const float b_o = bias[3 * H_ + hcol];

#pragma unroll
  for (int v = 0; v < 8; ++v) {
    const int brow = m0 + v + (hi << 3);
    const size_t s = (size_t)brow * H_ + hcol;
    float gi = acc0[v] + b_i;
    float gf = acc1[v] + b_f;
    float gg = acc2[v] + b_g;
    float go = acc3[v] + b_o;
    float cold = cs[s];
    float cn = sigmoidf_(gf) * cold + sigmoidf_(gi) * tanhf_(gg);
    float hv = sigmoidf_(go) * tanhf_(cn);
    cs[s] = cn;
    hs[s] = hv;
    hn[s] = f2bf(hv);
    if (outp) {
      outp[(size_t)brow * (T_ * 2 * H_) + (size_t)tout * (2 * H_) + dir * H_ + hcol] = hv;
    }
  }
}

// ---------------------------------------------------------------- host
extern "C" void kernel_launch(void* const* d_in, const int* in_sizes, int n_in,
                              void* d_out, int out_size, void* d_ws, size_t ws_size,
                              hipStream_t stream) {
  (void)in_sizes; (void)n_in; (void)out_size; (void)ws_size;

  const float* x     = (const float*)d_in[0];
  const float* h0    = (const float*)d_in[1];
  const float* c0    = (const float*)d_in[2];
  const float* Wih_f = (const float*)d_in[3];
  const float* Whh_f = (const float*)d_in[4];
  const float* bih_f = (const float*)d_in[5];
  const float* bhh_f = (const float*)d_in[6];
  const float* Wih_b = (const float*)d_in[7];
  const float* Whh_b = (const float*)d_in[8];
  const float* bih_b = (const float*)d_in[9];
  const float* bhh_b = (const float*)d_in[10];

  // ---- carve workspace
  char* w = (char*)d_ws;
  auto carve = [&](size_t bytes) -> void* {
    void* p = (void*)w;
    w += (bytes + 255) & ~(size_t)255;
    return p;
  };
  const int nX   = B_ * T_ * D_;       // 16,777,216
  const int nWih = L_ * G_ * D_;       // 2,097,152
  const int nWhh = L_ * G_ * H_;       // 2,097,152
  const int nBia = L_ * G_;            // 4096
  const size_t cellN = (size_t)B_ * H_;

  __bf16* x_bf    = (__bf16*)carve((size_t)nX * 2);
  __bf16* wih_fbf = (__bf16*)carve((size_t)nWih * 2);
  __bf16* whh_fbf = (__bf16*)carve((size_t)nWhh * 2);
  __bf16* wih_bbf = (__bf16*)carve((size_t)nWih * 2);
  __bf16* whh_bbf = (__bf16*)carve((size_t)nWhh * 2);
  float*  bsum_f  = (float*)carve((size_t)nBia * 4);
  float*  bsum_b  = (float*)carve((size_t)nBia * 4);
  float*  hs      = (float*)carve((size_t)L_ * 2 * cellN * 4);   // [l][dir][B][H]
  float*  cs      = (float*)carve((size_t)L_ * 2 * cellN * 4);
  __bf16* hbf     = (__bf16*)carve((size_t)L_ * 2 * 2 * cellN * 2); // [l][dir][buf][B][H]

  auto HS  = [&](int l, int d) { return hs + ((size_t)(l * 2 + d)) * cellN; };
  auto CS  = [&](int l, int d) { return cs + ((size_t)(l * 2 + d)) * cellN; };
  auto HBF = [&](int l, int d, int buf) {
    return hbf + ((size_t)((l * 2 + d) * 2 + buf)) * cellN;
  };

  // ---- prep
  cvt_f32_bf16_kernel<<<(nX + 255) / 256, 256, 0, stream>>>(x, x_bf, nX);
  cvt_f32_bf16_kernel<<<(nWih + 255) / 256, 256, 0, stream>>>(Wih_f, wih_fbf, nWih);
  cvt_f32_bf16_kernel<<<(nWhh + 255) / 256, 256, 0, stream>>>(Whh_f, whh_fbf, nWhh);
  cvt_f32_bf16_kernel<<<(nWih + 255) / 256, 256, 0, stream>>>(Wih_b, wih_bbf, nWih);
  cvt_f32_bf16_kernel<<<(nWhh + 255) / 256, 256, 0, stream>>>(Whh_b, whh_bbf, nWhh);
  bias_sum_kernel<<<(nBia + 255) / 256, 256, 0, stream>>>(bih_f, bhh_f, bsum_f, nBia);
  bias_sum_kernel<<<(nBia + 255) / 256, 256, 0, stream>>>(bih_b, bhh_b, bsum_b, nBia);
  {
    const int n = L_ * B_ * 2 * H_;
    init_state_kernel<<<(n + 255) / 256, 256, 0, stream>>>(h0, c0, hs, cs, hbf);
  }

  float* out = (float*)d_out;                              // (B, T, 2H)
  const size_t lstrW1 = (size_t)G_ * D_;                   // Wih layer stride
  const size_t lstrW2 = (size_t)G_ * H_;                   // Whh layer stride

  // ---- recurrence
  for (int t = 0; t < T_; ++t) {
    const int cur = t & 1, nxt = (t + 1) & 1;
    const int tb = T_ - 1 - t;

    // layer 0: x-part from input (row stride T*D), h-part recurrent
    lstm_step_kernel<<<32, 256, 0, stream>>>(
        x_bf + (size_t)t * D_, x_bf + (size_t)tb * D_, (long)T_ * D_, D_,
        wih_fbf, wih_bbf,
        HBF(0, 0, cur), HBF(0, 1, cur),
        whh_fbf, whh_bbf,
        bsum_f, bsum_b,
        CS(0, 0), CS(0, 1), HS(0, 0), HS(0, 1),
        HBF(0, 0, nxt), HBF(0, 1, nxt),
        nullptr, 0, 0);

    // layer 1: x-part = layer-0 new h, h-part recurrent; writes final output
    lstm_step_kernel<<<32, 256, 0, stream>>>(
        HBF(0, 0, nxt), HBF(0, 1, nxt), (long)H_, H_,
        wih_fbf + lstrW1, wih_bbf + lstrW1,
        HBF(1, 0, cur), HBF(1, 1, cur),
        whh_fbf + lstrW2, whh_bbf + lstrW2,
        bsum_f + G_, bsum_b + G_,
        CS(1, 0), CS(1, 1), HS(1, 0), HS(1, 1),
        HBF(1, 0, nxt), HBF(1, 1, nxt),
        out, t, tb);
  }

  // ---- final h, c outputs
  {
    float* outh = out + (size_t)B_ * T_ * 2 * H_;
    float* outc = outh + (size_t)L_ * B_ * 2 * H_;
    const int n = L_ * B_ * 2 * H_;
    finalize_kernel<<<(n + 255) / 256, 256, 0, stream>>>(hs, cs, outh, outc);
  }
}
